// SwinViTContextualLoss_34308198761205
// MI455X (gfx1250) — matrix operand order repaired and checked by reading, not compiled, not run on hardware
//
#include <hip/hip_runtime.h>
#include <hip/hip_bf16.h>

// CDNA5 (gfx1250) contextual-loss kernel.
// One workgroup per sample; Gram matrices via v_wmma_f32_16x16x32_f16.

typedef __attribute__((ext_vector_type(16))) _Float16 v16h;
typedef __attribute__((ext_vector_type(8)))  float    v8f;

#define LDK 72          // padded K stride in halves (bank-conflict avoidance)
#define KPAD 64         // K padded from C=48 to 64 (zeros)
#define CH   48
#define MM   512        // 8*8*8 neighborhood
#define VOX  262144     // 64*64*64
#define NSTR (48u * 262144u)

union FragU { v16h v; uint4 q[2]; };

// A-matrix 16x32 f16 fragment (documented CDNA5 wave32 layout):
// lanes 0-15: row M=lane, K = {0..7, 16..23}; lanes 16-31: row M=lane-16, K = {8..15, 24..31}
__device__ __forceinline__ v16h frag_a(const _Float16* A, int mt, int kb, int lane) {
  const int r = (mt << 4) + (lane & 15);
  const int o = (kb << 5) + ((lane & 16) ? 8 : 0);
  const uint4* p = (const uint4*)(A + r * LDK + o);
  FragU f;
  f.q[0] = p[0];   // K = o .. o+7
  f.q[1] = p[2];   // K = o+16 .. o+23
  return f.v;
}

// B-matrix 32x16 f16 fragment: lanes 0-15: col N=lane, K = {0..7, 16..23}; lanes 16-31: col N=lane-16, K = {8..15, 24..31}
__device__ __forceinline__ v16h frag_b(const _Float16* B, int pt, int kb, int lane) {
  const int p = (pt << 4) + (lane & 15);
  const int o = (kb << 5) + ((lane & 16) ? 8 : 0);
  const uint4* q = (const uint4*)(B + p * LDK + o);
  FragU f;
  f.q[0] = q[0];
  f.q[1] = q[2];
  return f.v;
}

__device__ __forceinline__ v8f wmma_f16(v16h a, v16h b, v8f c) {
  // (neg_a, A, neg_b, B, c_mod, C, reuse_a, reuse_b)
  return __builtin_amdgcn_wmma_f32_16x16x32_f16(false, a, false, b, (short)0, c, false, false);
}

__global__ __launch_bounds__(256) void ctx_loss_kernel(
    const float* __restrict__ x, const float* __restrict__ y,
    const int* __restrict__ hI, const int* __restrict__ wI, const int* __restrict__ dI,
    float* __restrict__ part)
{
  __shared__ __align__(16) _Float16 A_lds[MM * LDK];  // 73728 B
  __shared__ __align__(16) _Float16 B_lds[MM * LDK];  // 73728 B
  __shared__ float rowdmin[MM];
  __shared__ float rowrinv[MM];
  __shared__ float rowsum[MM];
  __shared__ float colmax[MM];
  __shared__ float ymu_part[192];
  __shared__ float ymu[CH];
  __shared__ float red[256];

  const int tid  = threadIdx.x;
  const int lane = tid & 31;
  const int wave = tid >> 5;
  const int s    = blockIdx.x;

  const int h0 = hI[s] - 4, w0 = wI[s] - 4, d0 = dI[s] - 4;
  const size_t corner = (size_t)h0 * 4096 + (size_t)w0 * 64 + (size_t)d0;

  // ---- y_mu[c] = mean over (n, 512 positions) of y neighborhood ----
  if (tid < 192) {
    const int c = tid >> 2, pp = tid & 3;
    float acc = 0.f;
    for (int j = 0; j < 256; ++j) {
      const int idx = pp * 256 + j;        // 0..1023 = n*512 + m
      const int n_ = idx >> 9, mm = idx & 511;
      const int mh = mm >> 6, mw = (mm >> 3) & 7, md = mm & 7;
      const size_t a = (size_t)n_ * NSTR + (size_t)c * VOX + corner
                     + (size_t)mh * 4096 + (size_t)mw * 64 + md;
      acc += y[a];
    }
    ymu_part[tid] = acc;
  }
  __syncthreads();
  if (tid < CH)
    ymu[tid] = (ymu_part[tid*4] + ymu_part[tid*4+1] + ymu_part[tid*4+2] + ymu_part[tid*4+3])
               * (1.0f / 1024.0f);
  __syncthreads();

  float loss_acc = 0.f;

  for (int n = 0; n < 2; ++n) {
    // ---- preprocess: center by y_mu, L2-normalize over C, f16 into LDS ----
    for (int mm = tid; mm < MM; mm += 256) {
      const int mh = mm >> 6, mw = (mm >> 3) & 7, md = mm & 7;
      const size_t sp = (size_t)n * NSTR + corner
                      + (size_t)mh * 4096 + (size_t)mw * 64 + md;
      float ssx = 0.f, ssy = 0.f;
      for (int c = 0; c < CH; ++c) {
        const float mu = ymu[c];
        const float xv = x[sp + (size_t)c * VOX] - mu; ssx += xv * xv;
        const float yv = y[sp + (size_t)c * VOX] - mu; ssy += yv * yv;
      }
      const float inx = 1.0f / fmaxf(sqrtf(ssx), 1e-12f);
      const float iny = 1.0f / fmaxf(sqrtf(ssy), 1e-12f);
      _Float16* Ar = &A_lds[mm * LDK];
      _Float16* Br = &B_lds[mm * LDK];
      for (int c = 0; c < CH; ++c) {
        const float mu = ymu[c];
        Ar[c] = (_Float16)((x[sp + (size_t)c * VOX] - mu) * inx);
        Br[c] = (_Float16)((y[sp + (size_t)c * VOX] - mu) * iny);
      }
      for (int c = CH; c < KPAD; ++c) { Ar[c] = (_Float16)0.f; Br[c] = (_Float16)0.f; }
    }
    __syncthreads();

    // ---- Pass A: dist row minima (wave owns tile-rows; deterministic) ----
    for (int mi = 0; mi < 4; ++mi) {
      const int mt = wave * 4 + mi;
      const v16h a0 = frag_a(A_lds, mt, 0, lane);
      const v16h a1 = frag_a(A_lds, mt, 1, lane);
      float rmin[8];
      #pragma unroll
      for (int g = 0; g < 8; ++g) rmin[g] = 1e30f;
      for (int pt = 0; pt < 32; ++pt) {
        const v16h b0 = frag_b(B_lds, pt, 0, lane);
        const v16h b1 = frag_b(B_lds, pt, 1, lane);
        v8f acc = {};
        acc = wmma_f16(a0, b0, acc);
        acc = wmma_f16(a1, b1, acc);
        #pragma unroll
        for (int g = 0; g < 8; ++g) rmin[g] = fminf(rmin[g], 1.0f - acc[g]);
      }
      const int rbase = mt * 16 + ((lane & 16) ? 8 : 0);
      #pragma unroll
      for (int g = 0; g < 8; ++g) {
        float v = rmin[g];
        v = fminf(v, __shfl_xor(v, 1));
        v = fminf(v, __shfl_xor(v, 2));
        v = fminf(v, __shfl_xor(v, 4));
        v = fminf(v, __shfl_xor(v, 8));
        if ((lane & 15) == 0) rowdmin[rbase + g] = v;   // single writer per row
      }
    }
    __syncthreads();
    for (int mm = tid; mm < MM; mm += 256)
      rowrinv[mm] = 1.0f / (rowdmin[mm] + 1e-5f);
    __syncthreads();

    // ---- Pass B: row sums of e' = exp((dmin - d) * rinv / BW), BW = 0.5 ----
    for (int mi = 0; mi < 4; ++mi) {
      const int mt = wave * 4 + mi;
      const v16h a0 = frag_a(A_lds, mt, 0, lane);
      const v16h a1 = frag_a(A_lds, mt, 1, lane);
      const int rbase = mt * 16 + ((lane & 16) ? 8 : 0);
      float dmin_g[8], rinv_g[8], rsum[8];
      #pragma unroll
      for (int g = 0; g < 8; ++g) {
        dmin_g[g] = rowdmin[rbase + g];
        rinv_g[g] = rowrinv[rbase + g];
        rsum[g]   = 0.f;
      }
      for (int pt = 0; pt < 32; ++pt) {
        const v16h b0 = frag_b(B_lds, pt, 0, lane);
        const v16h b1 = frag_b(B_lds, pt, 1, lane);
        v8f acc = {};
        acc = wmma_f16(a0, b0, acc);
        acc = wmma_f16(a1, b1, acc);
        #pragma unroll
        for (int g = 0; g < 8; ++g) {
          const float d = 1.0f - acc[g];
          rsum[g] += __expf((dmin_g[g] - d) * rinv_g[g] * 2.0f);
        }
      }
      #pragma unroll
      for (int g = 0; g < 8; ++g) {
        float v = rsum[g];
        v += __shfl_xor(v, 1);
        v += __shfl_xor(v, 2);
        v += __shfl_xor(v, 4);
        v += __shfl_xor(v, 8);
        if ((lane & 15) == 0) rowsum[rbase + g] = v;
      }
    }
    __syncthreads();

    // ---- Pass C: column max of cx = e'/rowsum (wave owns tile-cols) ----
    for (int pi = 0; pi < 4; ++pi) {
      const int pt = wave * 4 + pi;
      const v16h b0 = frag_b(B_lds, pt, 0, lane);
      const v16h b1 = frag_b(B_lds, pt, 1, lane);
      float cmax = 0.f;                      // cx >= 0
      for (int mt = 0; mt < 32; ++mt) {
        const v16h a0 = frag_a(A_lds, mt, 0, lane);
        const v16h a1 = frag_a(A_lds, mt, 1, lane);
        v8f acc = {};
        acc = wmma_f16(a0, b0, acc);
        acc = wmma_f16(a1, b1, acc);
        const int rbase = mt * 16 + ((lane & 16) ? 8 : 0);
        #pragma unroll
        for (int g = 0; g < 8; ++g) {
          const float d = 1.0f - acc[g];
          const float e = __expf((rowdmin[rbase + g] - d) * rowrinv[rbase + g] * 2.0f);
          cmax = fmaxf(cmax, e / rowsum[rbase + g]);
        }
      }
      cmax = fmaxf(cmax, __shfl_xor(cmax, 16));
      if (lane < 16) colmax[pt * 16 + lane] = cmax;
    }
    __syncthreads();

    // ---- reduce colmax -> mean over 512 -> -log(cx + eps) ----
    red[tid] = colmax[tid] + colmax[tid + 256];
    __syncthreads();
    for (int stp = 128; stp > 0; stp >>= 1) {
      if (tid < stp) red[tid] += red[tid + stp];
      __syncthreads();
    }
    if (tid == 0) {
      const float cxn = red[0] * (1.0f / 512.0f);
      loss_acc += -logf(cxn + 1e-5f);
    }
    __syncthreads();   // before LDS reuse for next n
  }

  if (tid == 0) part[s] = loss_acc;
}

__global__ void ctx_loss_finalize(const float* __restrict__ part, float* __restrict__ out, int S) {
  if (blockIdx.x == 0 && threadIdx.x == 0) {
    float acc = 0.f;
    for (int i = 0; i < S; ++i) acc += part[i];
    out[0] = acc / (2.0f * (float)S);   // mean over N=2 then mean over S
  }
}

extern "C" void kernel_launch(void* const* d_in, const int* in_sizes, int n_in,
                              void* d_out, int out_size, void* d_ws, size_t ws_size,
                              hipStream_t stream) {
  (void)n_in; (void)out_size; (void)ws_size;
  const float* x  = (const float*)d_in[0];
  const float* y  = (const float*)d_in[1];
  const int*   hI = (const int*)d_in[2];
  const int*   wI = (const int*)d_in[3];
  const int*   dI = (const int*)d_in[4];
  float* part = (float*)d_ws;
  float* out  = (float*)d_out;
  const int S = in_sizes[2];   // num samples (10)

  ctx_loss_kernel<<<S, 256, 0, stream>>>(x, y, hI, wI, dI, part);
  ctx_loss_finalize<<<1, 32, 0, stream>>>(part, out, S);
}
